// RNN_82703890252433
// MI455X (gfx1250) — matrix operand order repaired and checked by reading, not compiled, not run on hardware
//
#include <hip/hip_runtime.h>

// ---------------------------------------------------------------------------
// PredRNN++ (bidirectional CausalLSTM, 4 layers) forward on gfx1250.
// All 5x5 / 1x1 convs run as implicit GEMM on v_wmma_f32_16x16x32_f16.
// Activations are kept in f16, PIXEL-MAJOR layout [B][256][C] so GEMM B-tiles
// and epilogues are contiguous vector accesses; all math is f32 (WMMA accum).
// ---------------------------------------------------------------------------

typedef _Float16 h16;
typedef __attribute__((ext_vector_type(16))) _Float16 v16h;
typedef __attribute__((ext_vector_type(8)))  _Float16 v8h;
typedef __attribute__((ext_vector_type(4)))  _Float16 v4h;
typedef __attribute__((ext_vector_type(8)))  float    v8f;

#define PIXN 256          // 16x16 spatial
#define NE   32768        // B(2) * NH(64) * 256
#define ST   32768L       // per-time-step stride for C=64 seq buffers (halfs)

__device__ __forceinline__ float sigf(float x) { return 1.0f / (1.0f + expf(-x)); }

// ---------------------------------------------------------------------------
// Weight packing: OIHW f32 -> tap-major [25][Cout][Cin] f16 (rows contiguous
// in Cin so the GEMM A-tile stages with clean 128-bit vector loads).
// ---------------------------------------------------------------------------
__global__ __launch_bounds__(256)
void pack5_k(const float* __restrict__ w, h16* __restrict__ out, int Cout, int Cin) {
    long n = (long)Cout * Cin * 25;
    long i = (long)blockIdx.x * 256 + threadIdx.x;
    if (i >= n) return;
    int tap = (int)(i % 25);
    long r  = i / 25;
    int ci  = (int)(r % Cin);
    int co  = (int)(r / Cin);
    out[((long)tap * Cout + co) * Cin + ci] = (h16)w[i];
}

__global__ __launch_bounds__(256)
void pack1_k(const float* __restrict__ w, h16* __restrict__ out, int n) {
    int i = blockIdx.x * 256 + threadIdx.x;
    if (i < n) out[i] = (h16)w[i];
}

__global__ __launch_bounds__(256)
void fillh_k(h16* __restrict__ p, int n) {
    int i = blockIdx.x * 256 + threadIdx.x;
    if (i < n) p[i] = (h16)0.0f;
}

// Input cvt/transpose: fp32 [B,10,16,16,16] -> f16 [10][B][256][16]
__global__ __launch_bounds__(256)
void packseq_k(const float* __restrict__ src, h16* __restrict__ dst) {
    int i = blockIdx.x * 256 + threadIdx.x;
    if (i >= 2 * 10 * 16 * 256) return;
    int c   = i & 15;
    int pix = (i >> 4) & 255;
    int b   = (i >> 12) & 1;
    int t   = i >> 13;
    dst[i] = (h16)src[(((long)b * 10 + t) * 16 + c) * PIXN + pix];
}

// ---------------------------------------------------------------------------
// 5x5 SAME conv as implicit GEMM:
//   y[B,256,Cout] = im2col(x)[512, Cin*25] x W^T  (+bias, opt +=)
// Block = 256 threads = 8 waves tiling M=64 x N=32 (each wave one 16x16 tile).
// Pixel-major f16 activations make every staging access a contiguous vector
// load; predication is chunk-uniform (no per-lane exec churn).
// ---------------------------------------------------------------------------
__global__ __launch_bounds__(256)
void conv5_wmma(const h16* __restrict__ wp,   // [25][CoutTot][Cin] f16
                const float* __restrict__ bias,
                const h16* __restrict__ x,    // [B][256][Cin]
                h16* __restrict__ y,          // [B][256][CoutTot]
                int Cin, int CoutTot, int beta) {
    __shared__ __align__(16) h16 ldsA[64 * 32];
    __shared__ __align__(16) h16 ldsB[32 * 32];

    const int tid  = threadIdx.x;
    const int lane = tid & 31;
    const int wave = tid >> 5;
    const int mi = wave & 3;        // M sub-tile 0..3
    const int ni = wave >> 2;       // N sub-tile 0..1
    const int coBlock = blockIdx.x * 64;
    const int nBlock  = blockIdx.y * 32;

    // --- A staging geometry: thread owns one aligned 8-half chunk ---
    const int ai  = tid * 8;              // 0..2040
    const int am  = ai >> 5;              // row (M) 0..63
    const int akk = ai & 31;              // K offset within 32-wide step
    const h16* arow0 = wp + (long)(coBlock + am) * Cin;

    // --- B staging geometry: thread owns one aligned 4-half chunk ---
    const int bi  = tid * 4;              // 0..1020
    const int bn  = bi >> 5;              // col (N) 0..31
    const int bkk = bi & 31;
    const int bng = nBlock + bn;
    const int bb  = bng >> 8;
    const int bpy = (bng & 255) >> 4;
    const int bpx = bng & 15;
    const h16* xb0 = x + (long)bb * PIXN * Cin;

    // Fragment geometry (CDNA5 16-bit WMMA VGPR layouts, wave32):
    const int fr = lane & 15;            // M row (A) / N col (B) within tile
    const int kh = (lane >> 4) * 8;      // A: K half-offset pattern
    const int kb = (lane >> 4) * 16;     // unused; B also uses the kh pattern

    v8f acc = {};
    const long tapStride = (long)CoutTot * Cin;

    for (int tap = 0; tap < 25; ++tap) {
        const int dy = tap / 5 - 2;
        const int dx = tap % 5 - 2;
        const int py = bpy + dy;
        const int px = bpx + dx;
        const bool inImg = ((unsigned)py < 16u) & ((unsigned)px < 16u);
        const h16* xpix = xb0 + (long)(py * 16 + px) * Cin;

        for (int c0 = 0; c0 < Cin; c0 += 32) {
            // ---- stage A tile [64(M)][32(K)]: one b128 load per thread ----
            {
                v8h av = {};
                int k = c0 + akk;
                if (k + 8 <= Cin)
                    av = *(const v8h*)&arow0[(long)tap * tapStride + k];
                *(v8h*)&ldsA[ai] = av;
            }
            // ---- stage B^T tile [32(N)][32(K)]: contiguous 4-half chunk ----
            {
                v4h bv = {};
                int k = c0 + bkk;
                if (inImg && (k + 4 <= Cin))
                    bv = *(const v4h*)&xpix[k];
                *(v4h*)&ldsB[bi] = bv;
            }
            __syncthreads();

            // ---- build fragments ----
            const h16* ar = &ldsA[(mi * 16 + fr) * 32];
            v8h alo = *(const v8h*)&ar[kh];
            v8h ahi = *(const v8h*)&ar[16 + kh];
            const h16* br = &ldsB[(ni * 16 + fr) * 32];
            v8h blo = *(const v8h*)&br[kh];
            v8h bhi = *(const v8h*)&br[16 + kh];
            v16h a, b;
            #pragma unroll
            for (int q = 0; q < 8; ++q) {
                a[q] = alo[q]; a[q + 8] = ahi[q];
                b[q] = blo[q]; b[q + 8] = bhi[q];
            }
            acc = __builtin_amdgcn_wmma_f32_16x16x32_f16(
                false, a, false, b, (short)0, acc, false, false);
            __syncthreads();
        }
    }

    // ---- epilogue: 8 consecutive channels per lane -> one b128 store ----
    const int n  = lane & 15;
    const int mb = (lane >> 4) * 8;
    const int ng = nBlock + ni * 16 + n;
    const int b  = ng >> 8, pix = ng & 255;
    const int co = coBlock + mi * 16 + mb;
    const long off = ((long)b * PIXN + pix) * CoutTot + co;
    v8h ov;
    if (beta) {
        v8h old = *(const v8h*)&y[off];
        #pragma unroll
        for (int r = 0; r < 8; ++r)
            ov[r] = (h16)(acc[r] + bias[co + r] + (float)old[r]);
    } else {
        #pragma unroll
        for (int r = 0; r < 8; ++r)
            ov[r] = (h16)(acc[r] + bias[co + r]);
    }
    *(v8h*)&y[off] = ov;
}

// ---------------------------------------------------------------------------
// 1x1 conv (optionally over channel-concat of two inputs) as a single-wave
// WMMA GEMM:  y[B,256,Cout] = X[512,K] x W^T + bias.  C0 is a multiple of 16,
// so a lane's 16-K run never straddles the concat boundary; with pixel-major
// f16 the run is two contiguous b128 loads.
// ---------------------------------------------------------------------------
__global__ __launch_bounds__(32)
void conv1x1_wmma(const h16* __restrict__ wp, const float* __restrict__ bias,
                  const h16* __restrict__ x0, int C0,
                  const h16* __restrict__ x1, int C1,
                  h16* __restrict__ y, int Cout) {
    const int lane = threadIdx.x;
    const int co0 = blockIdx.x * 16;
    const int n0  = blockIdx.y * 16;
    const int K = C0 + C1;

    const int fr = lane & 15;
    const int kh = (lane >> 4) * 8;
    const int kb = (lane >> 4) * 16;
    const int ng = n0 + fr;
    const int b  = ng >> 8, pix = ng & 255;
    const h16* arow = wp + (long)(co0 + fr) * K;

    v8f acc = {};
    for (int k0 = 0; k0 < K; k0 += 32) {
        v8h alo = *(const v8h*)&arow[k0 + kb];
        v8h ahi = *(const v8h*)&arow[k0 + kb + 8];

        const int kbase = k0 + kb;
        const h16* xs = (kbase < C0)
            ? x0 + ((long)b * PIXN + pix) * C0 + kbase
            : x1 + ((long)b * PIXN + pix) * C1 + (kbase - C0);
        v8h blo = *(const v8h*)&xs[0];
        v8h bhi = *(const v8h*)&xs[8];

        v16h a, bf;
        #pragma unroll
        for (int q = 0; q < 8; ++q) {
            a[q]  = alo[q]; a[q + 8]  = ahi[q];
            bf[q] = blo[q]; bf[q + 8] = bhi[q];
        }
        acc = __builtin_amdgcn_wmma_f32_16x16x32_f16(
            false, a, false, bf, (short)0, acc, false, false);
    }

    const int n  = lane & 15;
    const int mb = (lane >> 4) * 8;
    const int ng2 = n0 + n;
    const int b2 = ng2 >> 8, pix2 = ng2 & 255;
    const int co = co0 + mb;
    v8h ov;
    #pragma unroll
    for (int r = 0; r < 8; ++r) ov[r] = (h16)(acc[r] + bias[co + r]);
    *(v8h*)&y[((long)b2 * PIXN + pix2) * Cout + co] = ov;
}

// ---------------------------------------------------------------------------
// Fused gate kernels.  All tensors are f16 pixel-major [B][256][C]; index
// i = b*16384 + pix*64 + ch addresses every C=64 tensor directly.
// Gx: C=448  Gh/G2: C=256  Gc/Gm: C=192  Gg: C=128
// ---------------------------------------------------------------------------
__global__ __launch_bounds__(256)
void gates1_k(const h16* __restrict__ Gx, const h16* __restrict__ Gh,
              const h16* __restrict__ Gc, const h16* __restrict__ c_in,
              h16* __restrict__ c_out) {
    int i = blockIdx.x * 256 + threadIdx.x;
    if (i >= NE) return;
    int ch = i & 63;
    long row = (long)(i >> 6);               // b*256 + pix
    const h16* gx = Gx + row * 448;
    const h16* gh = Gh + row * 256;
    const h16* gc = Gc + row * 192;
    float I = sigf((float)gx[ch] + (float)gh[ch] + (float)gc[ch]);
    float F = sigf((float)gx[64 + ch] + (float)gh[64 + ch] + (float)gc[64 + ch] + 1.0f);
    float G = tanhf((float)gx[128 + ch] + (float)gh[128 + ch] + (float)gc[128 + ch]);
    c_out[i] = (h16)(F * (float)c_in[i] + I * G);
}

__global__ __launch_bounds__(256)
void gates2_k(const h16* __restrict__ Gx, const h16* __restrict__ G2,
              const h16* __restrict__ Gm, h16* __restrict__ m_out) {
    int i = blockIdx.x * 256 + threadIdx.x;
    if (i >= NE) return;
    int ch = i & 63;
    long row = (long)(i >> 6);
    const h16* gx = Gx + row * 448;
    const h16* g2 = G2 + row * 256;          // split: i_c2, g_c2, f_c2, o_c
    const h16* gm = Gm + row * 192;          // split: i_m, f_m, m_m
    float I2 = sigf((float)gx[192 + ch] + (float)g2[ch] + (float)gm[ch]);
    float F2 = sigf((float)gx[256 + ch] + (float)g2[128 + ch] + (float)gm[64 + ch] + 1.0f);
    float G_ = tanhf((float)gx[320 + ch] + (float)g2[64 + ch]);
    m_out[i] = (h16)(F2 * tanhf((float)gm[128 + ch]) + I2 * G_);
}

__global__ __launch_bounds__(256)
void hout_k(const h16* __restrict__ Gx, const h16* __restrict__ Gh,
            const h16* __restrict__ G2, const h16* __restrict__ Gom,
            const h16* __restrict__ memb, h16* __restrict__ h_out) {
    int i = blockIdx.x * 256 + threadIdx.x;
    if (i >= NE) return;
    int ch = i & 63;
    long row = (long)(i >> 6);
    float o = tanhf((float)Gx[row * 448 + 384 + ch] + (float)Gh[row * 256 + 192 + ch] +
                    (float)G2[row * 256 + 192 + ch] + (float)Gom[i]);
    h_out[i] = (h16)(o * tanhf((float)memb[i]));
}

__global__ __launch_bounds__(256)
void ghu_k(const h16* __restrict__ Gg, const h16* __restrict__ z_in,
           h16* __restrict__ z_out) {
    int i = blockIdx.x * 256 + threadIdx.x;
    if (i >= NE) return;
    int ch = i & 63;
    long row = (long)(i >> 6);
    float u = sigf((float)Gg[row * 128 + 64 + ch]);
    z_out[i] = (h16)(u * tanhf((float)Gg[row * 128 + ch]) + (1.0f - u) * (float)z_in[i]);
}

// ---------------------------------------------------------------------------
// Final assembly: pixel-shuffle (r=4) even frames from fw_seq (fp32, original
// layout) and odd frames from gen (f16 [PRED][B][256][16]); out[B,10,1,64,64].
// ---------------------------------------------------------------------------
__global__ __launch_bounds__(256)
void assemble_k(const float* __restrict__ fw_seq, const h16* __restrict__ gen,
                float* __restrict__ out) {
    int i = blockIdx.x * 256 + threadIdx.x;
    if (i >= 2 * 10 * 64 * 64) return;
    int X = i & 63, Y = (i >> 6) & 63;
    int t = (i >> 12) % 10;
    int b = i / (10 * 4096);
    int c = (Y & 3) * 4 + (X & 3);
    int yy = Y >> 2, xx = X >> 2;
    float v;
    if ((t & 1) == 0)
        v = fw_seq[(((long)b * 10 + t) * 16 + c) * PIXN + yy * 16 + xx];
    else {
        int k = t >> 1;
        v = (float)gen[(((long)k * 2 + b) * PIXN + yy * 16 + xx) * 16 + c];
    }
    out[i] = v;
}

// ---------------------------------------------------------------------------
// Host orchestration.
// Input flattening (setup_inputs dict insertion order):
//   0: fw_seq  1: bw_seq
//   2..57   : params['fw'][0..3], 14 tensors/cell (Wx,bx,Wh,bh,Wc,bc,Wm,bm,
//             Wc2m,bc2m,Wom,bom,Wlast,blast)
//   58..113 : params['bw'][0..3]
//   114..117: ghu (Wz,bz,Wxg,bxg)
//   118..131: fuse h2,m2,h3,m3,h4,m4,gen (W,b each)
//   132,133 : last (W,b)
// ---------------------------------------------------------------------------
struct PCell {
    const h16 *Wx, *Wh, *Wc, *Wm, *Wc2m, *Wom, *Wlast;
    const float *bx, *bh, *bc, *bm, *bc2m, *bom, *blast;
    int cinX;
};

extern "C" void kernel_launch(void* const* d_in, const int* in_sizes, int n_in,
                              void* d_out, int out_size, void* d_ws, size_t ws_size,
                              hipStream_t stream) {
    (void)in_sizes; (void)n_in; (void)out_size; (void)ws_size;
    const float* fw_seq = (const float*)d_in[0];
    const float* bw_seq = (const float*)d_in[1];

    char* wsp = (char*)d_ws;
    auto take = [&](size_t bytes) -> void* {
        bytes = (bytes + 255) & ~(size_t)255;
        void* r = wsp; wsp += bytes; return r;
    };
    auto pk5 = [&](int idx, int Cout, int Cin) -> const h16* {
        h16* dst = (h16*)take((size_t)Cout * Cin * 25 * sizeof(h16));
        long n = (long)Cout * Cin * 25;
        pack5_k<<<dim3((unsigned)((n + 255) / 256)), dim3(256), 0, stream>>>(
            (const float*)d_in[idx], dst, Cout, Cin);
        return dst;
    };
    auto pk1 = [&](int idx, int n) -> const h16* {
        h16* dst = (h16*)take((size_t)n * sizeof(h16));
        pack1_k<<<dim3((n + 255) / 256), dim3(256), 0, stream>>>(
            (const float*)d_in[idx], dst, n);
        return dst;
    };
    auto mkcell = [&](int base, int cinX) -> PCell {
        PCell c; c.cinX = cinX;
        c.Wx   = pk5(base + 0, 448, cinX); c.bx   = (const float*)d_in[base + 1];
        c.Wh   = pk5(base + 2, 256, 64);   c.bh   = (const float*)d_in[base + 3];
        c.Wc   = pk5(base + 4, 192, 64);   c.bc   = (const float*)d_in[base + 5];
        c.Wm   = pk5(base + 6, 192, 64);   c.bm   = (const float*)d_in[base + 7];
        c.Wc2m = pk5(base + 8, 256, 64);   c.bc2m = (const float*)d_in[base + 9];
        c.Wom  = pk5(base + 10, 64, 64);   c.bom  = (const float*)d_in[base + 11];
        c.Wlast = pk1(base + 12, 64 * 128); c.blast = (const float*)d_in[base + 13];
        return c;
    };

    PCell fw[4], bw[4];
    for (int l = 0; l < 4; ++l) {
        fw[l] = mkcell(2 + l * 14, l ? 64 : 16);
        bw[l] = mkcell(58 + l * 14, l ? 64 : 16);
    }
    const h16* Wz  = pk5(114, 128, 64); const float* bz  = (const float*)d_in[115];
    const h16* Wxg = pk5(116, 128, 64); const float* bxg = (const float*)d_in[117];
    const h16* Ff[7]; const float* Fb[7];
    for (int f = 0; f < 7; ++f) {
        Ff[f] = pk1(118 + 2 * f, 64 * 128);
        Fb[f] = (const float*)d_in[119 + 2 * f];
    }
    const h16* WL = pk1(132, 16 * 64); const float* bL = (const float*)d_in[133];

    // -------- activation scratch (all f16, pixel-major [B][256][C]) --------
    auto hbuf = [&](size_t n) -> h16* { return (h16*)take(n * sizeof(h16)); };
    h16* Gx  = hbuf((size_t)2 * 256 * 448);
    h16* Gh  = hbuf((size_t)2 * 256 * 256);
    h16* Gc  = hbuf((size_t)2 * 256 * 192);
    h16* Gm  = hbuf((size_t)2 * 256 * 192);
    h16* G2  = hbuf((size_t)2 * 256 * 256);
    h16* Gom = hbuf(NE);
    h16* Gg  = hbuf((size_t)2 * 256 * 128);
    h16* memb = hbuf(NE);
    h16* zbuf = hbuf(NE);
    h16* cF = hbuf(NE); h16* cB = hbuf(NE);
    h16* hF = hbuf(NE); h16* hB = hbuf(NE);
    h16* mtF = hbuf(NE); h16* mtB = hbuf(NE);
    h16* zfs = hbuf(10 * ST); h16* mfs = hbuf(10 * ST);
    h16* zbs = hbuf(10 * ST); h16* mbs = hbuf(10 * ST);
    h16* h1f = hbuf(5 * ST); h16* m1f = hbuf(5 * ST);
    h16* h1b = hbuf(5 * ST); h16* m1b = hbuf(5 * ST);
    h16* h2f = hbuf(5 * ST); h16* m2f = hbuf(5 * ST);
    h16* h2b = hbuf(5 * ST); h16* m2b = hbuf(5 * ST);
    h16* h3f = hbuf(5 * ST); h16* h3b = hbuf(5 * ST);
    h16* hc2 = hbuf(5 * ST); h16* mc2 = hbuf(5 * ST);
    h16* hc3 = hbuf(5 * ST); h16* mc3 = hbuf(5 * ST);
    h16* hc4 = hbuf(5 * ST); h16* mc4 = hbuf(5 * ST);
    h16* genh = hbuf(5 * ST);
    h16* genb = hbuf((size_t)5 * 2 * 256 * 16);
    h16* xfh  = hbuf((size_t)10 * 2 * 256 * 16);
    h16* xbh  = hbuf((size_t)10 * 2 * 256 * 16);

    fillh_k<<<dim3(NE / 256), dim3(256), 0, stream>>>(zbuf, NE);
    packseq_k<<<dim3(320), dim3(256), 0, stream>>>(fw_seq, xfh);
    packseq_k<<<dim3(320), dim3(256), 0, stream>>>(bw_seq, xbh);

    // -------- launch helpers --------
    auto conv5 = [&](const h16* w, const float* b, const h16* x,
                     h16* y, int Cin, int Cout, int beta) {
        conv5_wmma<<<dim3(Cout / 64, 16), dim3(256), 0, stream>>>(
            w, b, x, y, Cin, Cout, beta);
    };
    auto conv1 = [&](const h16* w, const float* b, const h16* x0, int C0,
                     const h16* x1, int C1, h16* y, int Cout) {
        conv1x1_wmma<<<dim3(Cout / 16, 32), dim3(32), 0, stream>>>(
            w, b, x0, C0, x1 ? x1 : x0, C1, y, Cout);
    };
    auto cell = [&](const PCell& P, const h16* x,
                    const h16* h_in, h16* h_out,
                    const h16* c_in, h16* c_out,
                    const h16* m_in, h16* m_out) {
        conv5(P.Wx, P.bx, x, Gx, P.cinX, 448, 0);
        conv5(P.Wh, P.bh, h_in, Gh, 64, 256, 0);
        conv5(P.Wc, P.bc, c_in, Gc, 64, 192, 0);
        conv5(P.Wm, P.bm, m_in, Gm, 64, 192, 0);
        gates1_k<<<dim3(128), dim3(256), 0, stream>>>(Gx, Gh, Gc, c_in, c_out);
        conv5(P.Wc2m, P.bc2m, c_out, G2, 64, 256, 0);
        gates2_k<<<dim3(128), dim3(256), 0, stream>>>(Gx, G2, Gm, m_out);
        conv5(P.Wom, P.bom, m_out, Gom, 64, 64, 0);
        conv1(P.Wlast, P.blast, c_out, 64, m_out, 64, memb, 64);
        hout_k<<<dim3(128), dim3(256), 0, stream>>>(Gx, Gh, G2, Gom, memb, h_out);
    };
    auto ghu = [&](const h16* h, const h16* z_in, h16* z_out) {
        conv5(Wz,  bz,  z_in, Gg, 64, 128, 0);
        conv5(Wxg, bxg, h,    Gg, 64, 128, 1);
        ghu_k<<<dim3(128), dim3(256), 0, stream>>>(Gg, z_in, z_out);
    };

    // -------- layer 0: bidirectional scan over T=10 --------
    for (int t = 0; t < 10; ++t) {
        const h16* xf = xfh + (long)t * 2 * 256 * 16;
        const h16* xb = xbh + (long)t * 2 * 256 * 16;
        cell(fw[0], xf,
             t ? hF : zbuf, hF,
             t ? cF : zbuf, cF,
             t ? mfs + (t - 1) * ST : zbuf, mfs + t * ST);
        ghu(hF, t ? zfs + (t - 1) * ST : zbuf, zfs + t * ST);
        // NB: reference passes the freshly-updated forward cell state cF here
        cell(bw[0], xb,
             t ? hB : zbuf, hB,
             cF, cB,
             t ? mbs + (t - 1) * ST : zbuf, mbs + t * ST);
        ghu(hB, t ? zbs + (t - 1) * ST : zbuf, zbs + t * ST);
    }

    // -------- layer-2 fusion (even fw steps vs reversed even bw steps) -----
    for (int s = 0; s < 5; ++s) {
        conv1(Ff[0], Fb[0], zfs + (2 * s) * ST, 64, zbs + (8 - 2 * s) * ST, 64,
              hc2 + s * ST, 64);
        conv1(Ff[1], Fb[1], mfs + (2 * s) * ST, 64, mbs + (8 - 2 * s) * ST, 64,
              mc2 + s * ST, 64);
    }

    auto bilayer = [&](const PCell& Pf, const PCell& Pb,
                       const h16* hcf, const h16* mcf,
                       const h16* hcb, const h16* mcb,
                       h16* hof, h16* mof, h16* hob, h16* mob) {
        for (int s = 0; s < 5; ++s) {
            cell(Pf, hcf + s * ST,
                 s ? hof + (s - 1) * ST : zbuf, hof + s * ST,
                 s ? cF : zbuf, cF,
                 mcf + s * ST, mof ? mof + s * ST : mtF);
            cell(Pb, hcb + (4 - s) * ST,
                 s ? hob + (s - 1) * ST : zbuf, hob + s * ST,
                 s ? cB : zbuf, cB,
                 mcb + (4 - s) * ST, mob ? mob + s * ST : mtB);
        }
    };

    // -------- layer 1 --------
    bilayer(fw[1], bw[1], hc2, mc2, hc2, mc2, h1f, m1f, h1b, m1b);
    for (int s = 0; s < 5; ++s) {
        conv1(Ff[2], Fb[2], h1f + s * ST, 64, h1b + (4 - s) * ST, 64, hc3 + s * ST, 64);
        conv1(Ff[3], Fb[3], m1f + s * ST, 64, m1b + (4 - s) * ST, 64, mc3 + s * ST, 64);
    }
    // -------- layer 2 --------
    bilayer(fw[2], bw[2], hc3, mc3, hc3, mc3, h2f, m2f, h2b, m2b);
    for (int s = 0; s < 5; ++s) {
        conv1(Ff[4], Fb[4], h2f + s * ST, 64, h2b + (4 - s) * ST, 64, hc4 + s * ST, 64);
        conv1(Ff[5], Fb[5], m2f + s * ST, 64, m2b + (4 - s) * ST, 64, mc4 + s * ST, 64);
    }
    // -------- layer 3 (fw reuses layer-3 fused inputs, per reference) ------
    bilayer(fw[3], bw[3], hc3, mc3, hc4, mc4, h3f, nullptr, h3b, nullptr);
    for (int s = 0; s < 5; ++s)
        conv1(Ff[6], Fb[6], h3f + s * ST, 64, h3b + (4 - s) * ST, 64, genh + s * ST, 64);

    // -------- output head + frame assembly --------
    for (int k = 0; k < 5; ++k)
        conv1(WL, bL, genh + k * ST, 64, nullptr, 0, genb + (long)k * 2 * 256 * 16, 16);

    assemble_k<<<dim3(320), dim3(256), 0, stream>>>(fw_seq, genb, (float*)d_out);
}
